// DenseSAKELayer_48387101556869
// MI455X (gfx1250) — matrix operand order repaired and checked by reading, not compiled, not run on hardware
//
#include <hip/hip_runtime.h>

// DenseSAKELayer fused implementation for gfx1250 (MI455X).
//   node_pre:     per-node factorized GEMVs; per-j tensors stored TRANSPOSED
//                 ([b][h][N]) so the edge kernel loads them as b128 rows.
//   weights_prep: bf16-transposed weight tables (K padded to 64) for WMMA B.
//   edge_main:    one workgroup per (b,i): all per-edge work fused in LDS,
//                 3 edge GEMMs via v_wmma_f32_16x16x32_bf16, 3 softmaxes over j,
//                 coordinate + attention aggregation, final node MLPs.
// Assumes mask is binary. Requires ws_size >= ~1 MB.

#define BB    2
#define NNODE 512
#define FF    64
#define HH    64
#define NCC   32
#define NRBF_ 50
#define KPAD  64
#define FLD   68          // padded LDS row stride for filt (34 banks -> no row aliasing)
#define INF_  1e10f
#define EPS_  1e-5f
#define GAM_  10.0f

typedef __attribute__((ext_vector_type(16))) __bf16 v16bf;
typedef __attribute__((ext_vector_type(8)))  float  v8f;

union V16U { v16bf v; unsigned int u[8]; };

struct P {
  const float *h,*x,*mask,*W_dh,*b_dh,*mu,*W_f,*b_f,*W_sa,*W_ew,*b_ew,
              *W_pn1,*b_pn1,*W_pn2,*b_pn2,*W_nm1,*b_nm1,*W_nm2,*b_nm2,
              *W_cm1,*b_cm1,*W_cm2,*b_cm2;
  float *featjT,*feati,*saj,*sai,*ewjT,*ewi;   // featjT: [B][H][N]; ewjT: [B][NC][N]
  unsigned short *WfT,*Wcm1T,*Wew3T;
  float *out;
};

// ---- bf16 conversion: hardware v_cvt_pk_bf16_f32 when the builtin exists ---
#if defined(__has_builtin) && __has_builtin(__builtin_amdgcn_cvt_pk_bf16_f32)
typedef __bf16 v2bf __attribute__((ext_vector_type(2)));
__device__ __forceinline__ unsigned int pack_bf16(float lo, float hi){
  union { v2bf v; unsigned int u; } c;
  c.v = __builtin_amdgcn_cvt_pk_bf16_f32(lo, hi);
  return c.u;
}
__device__ __forceinline__ unsigned short f2bf(float f){
  return (unsigned short)(pack_bf16(f, 0.f) & 0xFFFFu);
}
#else
__device__ __forceinline__ unsigned short f2bf(float f){
  unsigned int x = __float_as_uint(f);
  x += 0x7FFFu + ((x >> 16) & 1u);          // RNE
  return (unsigned short)(x >> 16);
}
__device__ __forceinline__ unsigned int pack_bf16(float lo, float hi){
  return (unsigned int)f2bf(lo) | ((unsigned int)f2bf(hi) << 16);
}
#endif

__device__ __forceinline__ float bf2f(unsigned short s){
  return __uint_as_float(((unsigned int)s) << 16);
}
__device__ __forceinline__ float silu_f(float x){ return x / (1.0f + __expf(-x)); }

// ---- tanh: V_TANH_F32 TRANS op when exposed --------------------------------
__device__ __forceinline__ float tanh_fast(float x){
#if defined(__has_builtin) && __has_builtin(__builtin_amdgcn_tanhf)
  return __builtin_amdgcn_tanhf(x);
#else
  return tanhf(x);
#endif
}

__device__ __forceinline__ v8f wmma_bf16(v16bf a, v16bf b, v8f c){
  return __builtin_amdgcn_wmma_f32_16x16x32_bf16(false, a, false, b, (short)0, c, false, false);
}

// 16x32 bf16 A tile (rows row0..row0+15, K cols k0..k0+31) from LDS [row][k], ld elems.
__device__ __forceinline__ v16bf load_a16(const unsigned short* base, int row0, int ld,
                                          int k0, int lane){
  V16U t; int m = lane & 15, hf = lane >> 4;
  const unsigned short* p = base + (row0 + m) * ld + k0 + hf * 8;
  #pragma unroll
  for (int r = 0; r < 8; ++r){
    int kk = (r < 4) ? 2*r : 8 + 2*r;
    t.u[r] = *(const unsigned int*)(p + kk);
  }
  return t.v;
}

// 32x16 bf16 B tile (K k0..k0+31, N n0..n0+15) from LDS stored transposed [n][k].
__device__ __forceinline__ v16bf load_bT(const unsigned short* baseT, int n0, int ldk,
                                         int k0, int lane){
  V16U t; int n = n0 + (lane & 15), hf = lane >> 4;
  const unsigned short* p = baseT + n * ldk + k0 + hf * 8;
  #pragma unroll
  for (int r = 0; r < 8; ++r) t.u[r] = *(const unsigned int*)(p + ((r < 4) ? 2*r : 8 + 2*r));
  return t.v;
}

// RBF A tile built in registers: A[m][k] = exp(-GAMMA*(norm[jw+m]-mu[k])^2), k<NRBF else 0.
__device__ __forceinline__ v16bf mk_rbf(const float* mu_s, float nm, int k0, int lane){
  V16U t; int hf = lane >> 4;
  #pragma unroll
  for (int r = 0; r < 8; ++r){
    int kk = k0 + ((r < 4) ? 2*r : 8 + 2*r) + hf * 8;
    float v0 = 0.f, v1 = 0.f;
    if (kk     < NRBF_){ float d = nm - mu_s[kk];     v0 = __expf(-GAM_ * d * d); }
    if (kk + 1 < NRBF_){ float d = nm - mu_s[kk + 1]; v1 = __expf(-GAM_ * d * d); }
    t.u[r] = pack_bf16(v0, v1);
  }
  return t.v;
}

struct SM {
  unsigned short filt[NNODE * FLD];    // filtered[j][h] bf16, padded stride  68 KB
  unsigned short Wf[HH * KPAD];        // W_f  padded, [h][k]                  8 KB
  unsigned short Wcm1[HH * KPAD];      // W_cm1 [h'][k]                        8 KB
  unsigned short Wew3[NCC * KPAD];     // W_ew rows 2F.., [c][k]               4 KB
  float norm_s[NNODE];                 // norm; reused as total[] in phase2
  float s_sp[NNODE];                   // spatial logits -> probs
  float s_se[NNODE];                   // semantic logits -> probs
  float m_s[NNODE];                    // mask row
  float invn2[NNODE];                  // 1/(norm^2+EPS)
  float dvec[NNODE * 3];               // x[j]-x[i]
  float mu_s[KPAD], bf_s[KPAD], bcm1[KPAD], wcm2[KPAD], featiL[KPAD];
  float ewiL[NCC];
  float att[NCC * 3];
  float xacc[3], xi[3], sai1;
  float red[256];
  float part[8 * HH];
  float heagg[HH];
  float attn[NCC];
  float z[FF + 2 * HH];
  float tmp1[HH], tmp2[HH];
};

extern __shared__ unsigned char smraw[];

__device__ __forceinline__ float bred(float v, float* red, int t, int op){
  __syncthreads();
  red[t] = v; __syncthreads();
  #pragma unroll
  for (int s = 128; s > 0; s >>= 1){
    if (t < s) red[t] = op ? (red[t] + red[t + s]) : fmaxf(red[t], red[t + s]);
    __syncthreads();
  }
  return red[0];
}

__global__ void node_pre(P p){
  __shared__ float hs[FF];
  int n = blockIdx.x, t = threadIdx.x;
  int bb = n / NNODE, jl = n % NNODE;
  if (t < FF) hs[t] = p.h[n * FF + t];
  __syncthreads();
  if (t < HH){
    float a = 0.f, b = 0.f;
    for (int f = 0; f < FF; ++f){
      float hv = hs[f];
      a += hv * p.W_dh[f * HH + t];
      b += hv * p.W_dh[(FF + f) * HH + t];
    }
    p.featjT[((size_t)bb * HH + t) * NNODE + jl] = a;   // transposed [b][h][j]
    p.feati[n * HH + t] = b;
  } else if (t < HH + NCC){
    int c = t - HH; float a = 0.f, b = 0.f;
    for (int f = 0; f < FF; ++f){
      float hv = hs[f];
      a += hv * p.W_ew[f * NCC + c];
      b += hv * p.W_ew[(FF + f) * NCC + c];
    }
    p.ewjT[((size_t)bb * NCC + c) * NNODE + jl] = a;    // transposed [b][c][j]
    p.ewi[n * NCC + c] = b;
  } else if (t == HH + NCC){
    float a = 0.f, b = 0.f;
    for (int f = 0; f < FF; ++f){ float hv = hs[f]; a += hv * p.W_sa[f]; b += hv * p.W_sa[FF + f]; }
    p.saj[n] = a; p.sai[n] = b;
  }
}

__global__ void weights_prep(P p){
  int t = threadIdx.x;
  for (int idx = t; idx < HH * KPAD; idx += 256){
    int hc = idx / KPAD, k = idx % KPAD;
    p.WfT[idx]   = (k < NRBF_) ? f2bf(p.W_f[k * HH + hc]) : (unsigned short)0;
    p.Wcm1T[idx] = f2bf(p.W_cm1[k * HH + hc]);
  }
  for (int idx = t; idx < NCC * KPAD; idx += 256){
    int c = idx / KPAD, k = idx % KPAD;
    p.Wew3T[idx] = f2bf(p.W_ew[(2 * FF + k) * NCC + c]);
  }
}

__global__ void __launch_bounds__(256) edge_main(P p){
  SM& sm = *reinterpret_cast<SM*>(smraw);
  const int t = threadIdx.x;
  const int b = blockIdx.x / NNODE, i = blockIdx.x % NNODE;
  const int nb = b * NNODE, ni = nb + i;
  const int lane = t & 31, wave = t >> 5, m15 = lane & 15, hf = lane >> 4;
  const float bcm2v = p.b_cm2[0];

  // ---- stage weights / per-i vectors -------------------------------------
  for (int k = t; k < HH * KPAD; k += 256){ sm.Wf[k] = p.WfT[k]; sm.Wcm1[k] = p.Wcm1T[k]; }
  for (int k = t; k < NCC * KPAD; k += 256) sm.Wew3[k] = p.Wew3T[k];
  if (t < KPAD){
    sm.mu_s[t]   = (t < NRBF_) ? p.mu[t] : 0.f;
    sm.bf_s[t]   = p.b_f[t];
    sm.bcm1[t]   = p.b_cm1[t];
    sm.wcm2[t]   = p.W_cm2[t];
    sm.featiL[t] = p.feati[ni * HH + t] + p.b_dh[t];
  }
  if (t < NCC)     sm.ewiL[t] = p.ewi[ni * NCC + t] + p.b_ew[t];
  if (t < NCC * 3) sm.att[t] = 0.f;
  if (t < 3){ sm.xacc[t] = 0.f; sm.xi[t] = p.x[ni * 3 + t]; }
  if (t == 0) sm.sai1 = p.sai[ni];
  __syncthreads();

  // ---- pre-phase: per-edge scalars over all j ----------------------------
  const float sai_i = sm.sai1;
  #pragma unroll
  for (int jj = 0; jj < 2; ++jj){
    int j = t + jj * 256, jn = nb + j;
    float dx0 = p.x[jn*3+0] - sm.xi[0];
    float dx1 = p.x[jn*3+1] - sm.xi[1];
    float dx2 = p.x[jn*3+2] - sm.xi[2];
    float n2  = dx0*dx0 + dx1*dx1 + dx2*dx2;
    float nrm = sqrtf(fmaxf(n2, 0.f) + EPS_);
    float mkv = p.mask[(size_t)ni * NNODE + j];
    float eye = (j == i) ? 1.f : 0.f;
    sm.norm_s[j] = nrm;  sm.m_s[j] = mkv;
    sm.invn2[j]  = 1.f / (fmaxf(n2, 0.f) + 2.f * EPS_);   // norm^2 + EPS
    sm.dvec[j*3+0] = dx0; sm.dvec[j*3+1] = dx1; sm.dvec[j*3+2] = dx2;
    sm.s_sp[j] = -(nrm + INF_ * eye + (1.f - mkv) * INF_);
    float sl = p.saj[jn] + sai_i;
    sm.s_se[j] = silu_f(sl) - INF_ * eye + (mkv - 1.f) * INF_;
  }
  __syncthreads();

  // Hoist loop-invariant Wew3 B tiles (4 tiles = 32 VGPRs).
  v16bf bew[4];
  #pragma unroll
  for (int nt = 0; nt < 2; ++nt){
    bew[nt*2+0] = load_bT(sm.Wew3, nt*16, KPAD, 0,  lane);
    bew[nt*2+1] = load_bT(sm.Wew3, nt*16, KPAD, 32, lane);
  }

  // ---- main edge loop: each wave owns 16 j rows per tile-group -----------
  float wacc[2][3] = {{0,0,0},{0,0,0}};
  for (int g = 0; g < 4; ++g){
    const int jw = g * 128 + wave * 16;

    // GEMM1: rbf @ W_f  -> filtered (bf16 into LDS)
    float nm = sm.norm_s[jw + m15];
    v16bf ra0 = mk_rbf(sm.mu_s, nm, 0, lane);
    v16bf ra1 = mk_rbf(sm.mu_s, nm, 32, lane);
    #pragma unroll
    for (int nt = 0; nt < 4; ++nt){
      v8f acc = {};
      acc = wmma_bf16(ra0, load_bT(sm.Wf, nt*16, KPAD, 0,  lane), acc);
      acc = wmma_bf16(ra1, load_bT(sm.Wf, nt*16, KPAD, 32, lane), acc);
      int ncol = nt * 16 + m15;
      float bfv = sm.bf_s[ncol], fi = sm.featiL[ncol];
      // featjT row: 8 consecutive j -> two b128 loads
      const float4* f4 = (const float4*)&p.featjT[((size_t)b * HH + ncol) * NNODE + jw + hf * 8];
      float4 fA = f4[0], fB = f4[1];
      float fr[8] = {fA.x, fA.y, fA.z, fA.w, fB.x, fB.y, fB.z, fB.w};
      #pragma unroll
      for (int r = 0; r < 8; ++r){
        int j = jw + hf * 8 + r;                   // D rows: lo half M=r, hi half M=8+r
        sm.filt[j * FLD + ncol] = f2bf((fr[r] + fi) * (acc[r] + bfv));
      }
    }
    // LDS is serviced in-order per wave -> our loads below see our stores.
    v16bf fa0 = load_a16(sm.filt, jw, FLD, 0,  lane);
    v16bf fa1 = load_a16(sm.filt, jw, FLD, 32, lane);

    // GEMM2: filtered @ W_cm1 -> silu -> @W_cm2 -> _h_e -> coordinate update
    float prt[8] = {0,0,0,0,0,0,0,0};
    #pragma unroll
    for (int nt = 0; nt < 4; ++nt){
      v8f acc = {};
      acc = wmma_bf16(fa0, load_bT(sm.Wcm1, nt*16, KPAD, 0,  lane), acc);
      acc = wmma_bf16(fa1, load_bT(sm.Wcm1, nt*16, KPAD, 32, lane), acc);
      int ncol = nt * 16 + m15;
      float bc = sm.bcm1[ncol], wc = sm.wcm2[ncol];
      #pragma unroll
      for (int r = 0; r < 8; ++r) prt[r] += silu_f(acc[r] + bc) * wc;
    }
    #pragma unroll
    for (int off = 1; off < 16; off <<= 1){
      #pragma unroll
      for (int r = 0; r < 8; ++r) prt[r] += __shfl_xor(prt[r], off, 32);
    }
    if (m15 == 0){
      #pragma unroll
      for (int r = 0; r < 8; ++r){
        int j = jw + hf * 8 + r;
        float he = (prt[r] + bcm2v) * sm.m_s[j];
        atomicAdd(&sm.xacc[0], sm.dvec[j*3+0] * he);
        atomicAdd(&sm.xacc[1], sm.dvec[j*3+1] * he);
        atomicAdd(&sm.xacc[2], sm.dvec[j*3+2] * he);
      }
    }

    // GEMM3: filtered @ W_ew3 -> tanh -> att accumulation
    #pragma unroll
    for (int nt = 0; nt < 2; ++nt){
      v8f acc = {};
      acc = wmma_bf16(fa0, bew[nt*2+0], acc);
      acc = wmma_bf16(fa1, bew[nt*2+1], acc);
      int c = nt * 16 + m15;
      float ei = sm.ewiL[c];
      const float4* e4 = (const float4*)&p.ewjT[((size_t)b * NCC + c) * NNODE + jw + hf * 8];
      float4 eA = e4[0], eB = e4[1];
      float er[8] = {eA.x, eA.y, eA.z, eA.w, eB.x, eB.y, eB.z, eB.w};
      #pragma unroll
      for (int r = 0; r < 8; ++r){
        int j = jw + hf * 8 + r;
        float wv = tanh_fast(acc[r] + er[r] + ei);
        float s  = wv * sm.m_s[j] * sm.invn2[j];
        wacc[nt][0] += s * sm.dvec[j*3+0];
        wacc[nt][1] += s * sm.dvec[j*3+1];
        wacc[nt][2] += s * sm.dvec[j*3+2];
      }
    }
  }
  #pragma unroll
  for (int nt = 0; nt < 2; ++nt){
    int c = nt * 16 + m15;
    atomicAdd(&sm.att[c*3+0], wacc[nt][0]);
    atomicAdd(&sm.att[c*3+1], wacc[nt][1]);
    atomicAdd(&sm.att[c*3+2], wacc[nt][2]);
  }
  __syncthreads();

  // ---- phase 2: triple softmax over j + h_e aggregation ------------------
  float M1 = bred(fmaxf(sm.s_sp[t], sm.s_sp[t+256]), sm.red, t, 0);
  float e0 = __expf(sm.s_sp[t] - M1), e1 = __expf(sm.s_sp[t+256] - M1);
  float S1 = bred(e0 + e1, sm.red, t, 1);
  float iv = 1.f / S1; sm.s_sp[t] = e0 * iv; sm.s_sp[t+256] = e1 * iv;

  float M2 = bred(fmaxf(sm.s_se[t], sm.s_se[t+256]), sm.red, t, 0);
  float f0 = __expf(sm.s_se[t] - M2), f1 = __expf(sm.s_se[t+256] - M2);
  float S2 = bred(f0 + f1, sm.red, t, 1);
  iv = 1.f / S2; sm.s_se[t] = f0 * iv; sm.s_se[t+256] = f1 * iv;

  float tl0 = sm.s_se[t]     * sm.s_sp[t]     + (sm.m_s[t]     - 1.f) * INF_;
  float tl1 = sm.s_se[t+256] * sm.s_sp[t+256] + (sm.m_s[t+256] - 1.f) * INF_;
  float M3 = bred(fmaxf(tl0, tl1), sm.red, t, 0);
  float g0 = __expf(tl0 - M3), g1 = __expf(tl1 - M3);
  float S3 = bred(g0 + g1, sm.red, t, 1);
  iv = 1.f / S3;
  sm.norm_s[t] = g0 * iv; sm.norm_s[t+256] = g1 * iv;   // norm_s reused as total[]
  __syncthreads();

  // h_e_agg: 8 j-groups x 32 threads, each thread handles 2 bf16 columns per u32 load
  {
    int grp = t >> 5, hp = (t & 31) * 2;
    float s0 = 0.f, s1 = 0.f;
    for (int j = grp * 64; j < grp * 64 + 64; ++j){
      unsigned int uu = *(const unsigned int*)&sm.filt[j * FLD + hp];
      float w = sm.norm_s[j] * sm.m_s[j];
      s0 += w * bf2f((unsigned short)(uu & 0xFFFFu));
      s1 += w * bf2f((unsigned short)(uu >> 16));
    }
    sm.part[grp * HH + hp]     = s0;
    sm.part[grp * HH + hp + 1] = s1;
  }
  __syncthreads();
  if (t < HH){
    float a = 0.f;
    #pragma unroll
    for (int q = 0; q < 8; ++q) a += sm.part[q * HH + t];
    sm.heagg[t] = a;
  }

  // ---- phase 3: per-node MLPs --------------------------------------------
  if (t < NCC){
    float s = sm.att[t*3]*sm.att[t*3] + sm.att[t*3+1]*sm.att[t*3+1] + sm.att[t*3+2]*sm.att[t*3+2];
    sm.attn[t] = sqrtf(fmaxf(s, 0.f) + EPS_);
  }
  __syncthreads();
  if (t < HH){
    float a = p.b_pn1[t];
    for (int c = 0; c < NCC; ++c) a += sm.attn[c] * p.W_pn1[c * HH + t];
    sm.tmp1[t] = silu_f(a);
  }
  __syncthreads();
  if (t < HH){
    float a = p.b_pn2[t];
    for (int k = 0; k < HH; ++k) a += sm.tmp1[k] * p.W_pn2[k * HH + t];
    sm.z[2 * HH + t] = a;                 // emb
    sm.z[t]          = p.h[ni * FF + t];  // h
    sm.z[FF + t]     = sm.heagg[t];       // h_e_agg
  }
  __syncthreads();
  if (t < HH){
    float a = p.b_nm1[t];
    for (int k = 0; k < FF + 2 * HH; ++k) a += sm.z[k] * p.W_nm1[k * HH + t];
    sm.tmp2[t] = silu_f(a);
  }
  __syncthreads();
  if (t < HH){
    float a = p.b_nm2[t];
    for (int k = 0; k < HH; ++k) a += sm.tmp2[k] * p.W_nm2[k * HH + t];
    p.out[ni * HH + t] = a;
  }
  if (t < 3) p.out[BB * NNODE * HH + ni * 3 + t] = sm.xacc[t] + sm.xi[t];
}

extern "C" void kernel_launch(void* const* d_in, const int* in_sizes, int n_in,
                              void* d_out, int out_size, void* d_ws, size_t ws_size,
                              hipStream_t stream){
  (void)in_sizes; (void)n_in; (void)out_size; (void)ws_size;
  P p;
  p.h     = (const float*)d_in[0];  p.x     = (const float*)d_in[1];
  p.mask  = (const float*)d_in[2];  p.W_dh  = (const float*)d_in[3];
  p.b_dh  = (const float*)d_in[4];  p.mu    = (const float*)d_in[5];
  p.W_f   = (const float*)d_in[6];  p.b_f   = (const float*)d_in[7];
  p.W_sa  = (const float*)d_in[8];  p.W_ew  = (const float*)d_in[9];
  p.b_ew  = (const float*)d_in[10]; p.W_pn1 = (const float*)d_in[11];
  p.b_pn1 = (const float*)d_in[12]; p.W_pn2 = (const float*)d_in[13];
  p.b_pn2 = (const float*)d_in[14]; p.W_nm1 = (const float*)d_in[15];
  p.b_nm1 = (const float*)d_in[16]; p.W_nm2 = (const float*)d_in[17];
  p.b_nm2 = (const float*)d_in[18]; p.W_cm1 = (const float*)d_in[19];
  p.b_cm1 = (const float*)d_in[20]; p.W_cm2 = (const float*)d_in[21];
  p.b_cm2 = (const float*)d_in[22];

  char* w = (char*)d_ws; size_t off = 0;
  auto alloc = [&](size_t bytes) -> void* {
    void* r = w + off; off = (off + bytes + 255) & ~(size_t)255; return r;
  };
  const int NT = BB * NNODE;
  p.featjT = (float*)alloc((size_t)NT * HH  * 4);
  p.feati  = (float*)alloc((size_t)NT * HH  * 4);
  p.saj    = (float*)alloc((size_t)NT * 4);
  p.sai    = (float*)alloc((size_t)NT * 4);
  p.ewjT   = (float*)alloc((size_t)NT * NCC * 4);
  p.ewi    = (float*)alloc((size_t)NT * NCC * 4);
  p.WfT    = (unsigned short*)alloc((size_t)HH  * KPAD * 2);
  p.Wcm1T  = (unsigned short*)alloc((size_t)HH  * KPAD * 2);
  p.Wew3T  = (unsigned short*)alloc((size_t)NCC * KPAD * 2);
  p.out    = (float*)d_out;

  node_pre<<<dim3(NT), dim3(128), 0, stream>>>(p);
  weights_prep<<<dim3(1), dim3(256), 0, stream>>>(p);
  edge_main<<<dim3(NT), dim3(256), sizeof(SM), stream>>>(p);
}
